// GCN_63471026700330
// MI455X (gfx1250) — matrix operand order repaired and checked by reading, not compiled, not run on hardware
//
#include <hip/hip_runtime.h>
#include <stdint.h>

typedef __attribute__((ext_vector_type(16))) __bf16 v16bf;
typedef __attribute__((ext_vector_type(8)))  __bf16 v8bf;
typedef __attribute__((ext_vector_type(4)))  __bf16 v4bf;
typedef __attribute__((ext_vector_type(8)))  float  v8f;

#define LDS_STRIDE 40   // 32 bf16 payload + 8 pad -> 80B row stride, conflict-free b128 frag loads

// ---- CDNA5 async global->LDS copy (ASYNCcnt) ----------------------------------
__device__ __forceinline__ void async_b128(unsigned ldsoff, const void* g) {
  asm volatile("global_load_async_to_lds_b128 %0, %1, off"
               :: "v"(ldsoff), "v"((unsigned long long)(uintptr_t)g)
               : "memory");
}
__device__ __forceinline__ void async_b128_nt(unsigned ldsoff, const void* g) {
  asm volatile("global_load_async_to_lds_b128 %0, %1, off th:TH_LOAD_NT"
               :: "v"(ldsoff), "v"((unsigned long long)(uintptr_t)g)
               : "memory");
}
__device__ __forceinline__ void wait_async0() {
  asm volatile("s_wait_asynccnt 0x0" ::: "memory");
}
__device__ __forceinline__ void wait_async_le4() {
  asm volatile("s_wait_asynccnt 0x4" ::: "memory");
}

// A-fragment, contiguous K (matches frag_b):
//   lanes 0-15 : row M=lane,    halves 0..15 = K0..15
//   lanes16-31 : row M=lane-16, halves 0..15 = K16..31
__device__ __forceinline__ v16bf frag_a(const __bf16* s, int row0, int lane) {
  int m  = lane & 15;
  int hi = lane >> 4;
  const __bf16* p = s + (row0 + m) * LDS_STRIDE;
  v8bf lo = *(const v8bf*)(p + (hi ? 16 : 0));
  v8bf h  = *(const v8bf*)(p + (hi ? 24 : 8));
  return __builtin_shufflevector(lo, h, 0,1,2,3,4,5,6,7,8,9,10,11,12,13,14,15);
}

// B-fragment: lane holds column n=lane&15; lanes 0-15 K=0..15, lanes 16-31 K=16..31
// (contiguous K) -> read from Bt stored N-major (row n, contiguous k) in LDS.
__device__ __forceinline__ v16bf frag_b(const __bf16* s, int row0, int lane) {
  int n  = lane & 15;
  int hi = lane >> 4;
  const __bf16* p = s + (row0 + n) * LDS_STRIDE + (hi ? 16 : 0);
  v8bf lo = *(const v8bf*)(p);
  v8bf h  = *(const v8bf*)(p + 8);
  return __builtin_shufflevector(lo, h, 0,1,2,3,4,5,6,7,8,9,10,11,12,13,14,15);
}

// C = A(MxK) * Bt^T (Bt is N x K bf16)  [+bias] [relu], out f32 or bf16.
// AMODE 0: A f32, register staging (fallback).  AMODE 1: A bf16, async DMA to LDS.
// A_NT: non-temporal hint on A-side async loads (streaming operand, protect L2).
template<int AMODE, bool RELU, bool HAS_BIAS, bool OUT_BF16, bool A_NT>
__global__ __launch_bounds__(256) void gemm_wmma(
    const void* __restrict__ Aptr, const __bf16* __restrict__ Bt,
    const float* __restrict__ bias, void* __restrict__ Cptr,
    int M, int N, int K, int lda, int ldb)
{
  __shared__ __bf16 sA[3][128 * LDS_STRIDE];
  __shared__ __bf16 sB[3][128 * LDS_STRIDE];

  const int tid  = threadIdx.x;
  const int lane = tid & 31;
  const int wave = tid >> 5;
  const int wm   = (wave >> 2) * 64;   // wave M offset (0/64)
  const int wn   = (wave & 3) * 32;    // wave N offset (0/32/64/96)
  const int m0   = blockIdx.y * 128;
  const int n0   = blockIdx.x * 128;

  const float*  Af = (const float*)Aptr;
  const __bf16* Ah = (const __bf16*)Aptr;

  const int arf = tid >> 3;        // f32 A loader: rows 0..31 (+32*i), float4 cols
  const int acf = (tid & 7) << 2;
  const int arh = tid >> 2;        // bf16 loader: rows 0..63 (+64*i), 8-elem cols
  const int ach = (tid & 3) << 3;

  const int numK     = (K + 31) >> 5;
  const int numKbody = K >> 5;         // full 32-wide tiles (tail peeled)

  v8f acc[4][2] = {};

  auto compute = [&](int buf) {
    const __bf16* As = sA[buf];
    const __bf16* Bs = sB[buf];
    v16bf fa[4], fb[2];
#pragma unroll
    for (int i = 0; i < 4; ++i) fa[i] = frag_a(As, wm + 16 * i, lane);
#pragma unroll
    for (int j = 0; j < 2; ++j) fb[j] = frag_b(Bs, wn + 16 * j, lane);
#pragma unroll
    for (int i = 0; i < 4; ++i)
#pragma unroll
      for (int j = 0; j < 2; ++j)
        acc[i][j] = __builtin_amdgcn_wmma_f32_16x16x32_bf16(
            false, fa[i], false, fb[j], (short)0, acc[i][j], false, false);
  };

  if (AMODE == 1) {
    // ------- async DMA path, 3-buffer / depth-2 pipeline, branch-free body -------
    // Body tiles: all K in-bounds. OOB rows (m>=M / n>=N) are CLAMPED to row 0:
    // they load garbage that only reaches output rows/cols masked at the store.
    // => exactly 4 async ops per thread per tile => partial ASYNCcnt waits are exact.
    auto tile_fast = [&](int kt, int buf) {
      const int kb = kt << 5;
#pragma unroll
      for (int i = 0; i < 2; ++i) {
        int m = m0 + arh + 64 * i;  m = (m < M) ? m : 0;
        unsigned lp = (unsigned)(uintptr_t)&sA[buf][(arh + 64 * i) * LDS_STRIDE + ach];
        const void* g = Ah + (size_t)m * lda + (kb + ach);
        if (A_NT) async_b128_nt(lp, g); else async_b128(lp, g);
      }
#pragma unroll
      for (int i = 0; i < 2; ++i) {
        int n = n0 + arh + 64 * i;  n = (n < N) ? n : 0;
        unsigned lp = (unsigned)(uintptr_t)&sB[buf][(arh + 64 * i) * LDS_STRIDE + ach];
        async_b128(lp, Bt + (size_t)n * ldb + (kb + ach));
      }
    };
    // K-tail tile: guarded, zero-fill OOB chunks (divergent OK, followed by wait 0)
    auto tile_tail = [&](int kt, int buf) {
      const int kb = kt << 5;
#pragma unroll
      for (int i = 0; i < 2; ++i) {
        int m = m0 + arh + 64 * i;
        int k = kb + ach;
        __bf16* lp = &sA[buf][(arh + 64 * i) * LDS_STRIDE + ach];
        if (m < M && k < K) {
          if (A_NT) async_b128_nt((unsigned)(uintptr_t)lp, Ah + (size_t)m * lda + k);
          else      async_b128((unsigned)(uintptr_t)lp,    Ah + (size_t)m * lda + k);
        } else { uint4 z = {0u, 0u, 0u, 0u}; *(uint4*)lp = z; }
      }
#pragma unroll
      for (int i = 0; i < 2; ++i) {
        int n = n0 + arh + 64 * i;
        int k = kb + ach;
        __bf16* lp = &sB[buf][(arh + 64 * i) * LDS_STRIDE + ach];
        if (n < N && k < K) async_b128((unsigned)(uintptr_t)lp, Bt + (size_t)n * ldb + k);
        else { uint4 z = {0u, 0u, 0u, 0u}; *(uint4*)lp = z; }
      }
    };

    int cb = 0;
    if (numKbody > 0) {
      tile_fast(0, 0);
      if (numKbody > 1) { tile_fast(1, 1); wait_async_le4(); }
      else              { wait_async0(); }
      __syncthreads();
      for (int kt = 0; kt < numKbody; ++kt) {
        if (kt + 2 < numKbody) {
          int pfb = cb + 2; if (pfb >= 3) pfb -= 3;
          tile_fast(kt + 2, pfb);
          compute(cb);
          wait_async_le4();     // tile kt+1 complete; tile kt+2 may stay in flight
        } else {
          compute(cb);
          wait_async0();
        }
        __syncthreads();
        cb = (cb + 1 == 3) ? 0 : cb + 1;
      }
    }
    if (numKbody < numK) {      // K-tail (e.g. K=10000 -> 16 leftover columns)
      tile_tail(numKbody, cb);
      wait_async0();
      __syncthreads();
      compute(cb);
    }
  } else {
    // ---------------- f32-A fallback: register staging + LDS convert ----------------
    float4 aregf[4];
    uint4  bregh[2];
    auto gload = [&](int kt) {
      const int kb = kt << 5;
#pragma unroll
      for (int i = 0; i < 4; ++i) {
        int m = m0 + arf + 32 * i;
        int k = kb + acf;
        float4 z = {0.f, 0.f, 0.f, 0.f};
        aregf[i] = (m < M && k < K) ? *(const float4*)(Af + (size_t)m * lda + k) : z;
      }
#pragma unroll
      for (int i = 0; i < 2; ++i) {
        int n = n0 + arh + 64 * i;
        int k = kb + ach;
        uint4 z = {0u, 0u, 0u, 0u};
        bregh[i] = (n < N && k < K) ? *(const uint4*)(Bt + (size_t)n * ldb + k) : z;
      }
    };
    auto sstore = [&](int buf) {
#pragma unroll
      for (int i = 0; i < 4; ++i) {
        v4bf v;
        v[0] = (__bf16)aregf[i].x; v[1] = (__bf16)aregf[i].y;
        v[2] = (__bf16)aregf[i].z; v[3] = (__bf16)aregf[i].w;
        *(v4bf*)&sA[buf][(arf + 32 * i) * LDS_STRIDE + acf] = v;
      }
#pragma unroll
      for (int i = 0; i < 2; ++i)
        *(uint4*)&sB[buf][(arh + 64 * i) * LDS_STRIDE + ach] = bregh[i];
    };
    gload(0);
    sstore(0);
    __syncthreads();
    for (int kt = 0; kt < numK; ++kt) {
      if (kt + 1 < numK) gload(kt + 1);
      compute(kt & 1);
      if (kt + 1 < numK) sstore((kt + 1) & 1);
      __syncthreads();
    }
  }

  // C layout: VGPR v -> row = base + v + (lane>=16 ? 8 : 0); col = base + (lane&15)
  float*  Cf = (float*)Cptr;
  __bf16* Ch = (__bf16*)Cptr;
#pragma unroll
  for (int i = 0; i < 4; ++i) {
#pragma unroll
    for (int j = 0; j < 2; ++j) {
      int c  = n0 + wn + 16 * j + (lane & 15);
      int rb = m0 + wm + 16 * i + (lane >> 4) * 8;
      if (c < N) {
        float bv = HAS_BIAS ? bias[c] : 0.f;
#pragma unroll
        for (int v = 0; v < 8; ++v) {
          int r = rb + v;
          if (r < M) {
            float val = acc[i][j][v] + bv;
            if (RELU) val = fmaxf(val, 0.f);
            if (OUT_BF16) Ch[(size_t)r * N + c] = (__bf16)val;
            else          Cf[(size_t)r * N + c] = val;
          }
        }
      }
    }
  }
}

// out (C x R, bf16) = transpose(in (R x C, f32))
__global__ __launch_bounds__(256) void transpose_f32_bf16(
    const float* __restrict__ in, __bf16* __restrict__ out, int R, int C)
{
  __shared__ __bf16 t[32][33];
  int c  = blockIdx.x * 32 + threadIdx.x;
  int r0 = blockIdx.y * 32;
#pragma unroll
  for (int ii = 0; ii < 4; ++ii) {
    int i = threadIdx.y + ii * 8;
    int r = r0 + i;
    t[i][threadIdx.x] = (r < R && c < C) ? (__bf16)in[(size_t)r * C + c] : (__bf16)0.0f;
  }
  __syncthreads();
  int rr = r0 + threadIdx.x;
#pragma unroll
  for (int ii = 0; ii < 4; ++ii) {
    int i  = threadIdx.y + ii * 8;
    int cc = blockIdx.x * 32 + i;
    if (cc < C && rr < R) out[(size_t)cc * R + rr] = t[threadIdx.x][i];
  }
}

// flat f32 -> bf16 convert (n multiple of 8)
__global__ __launch_bounds__(256) void convert_f32_bf16(
    const float* __restrict__ in, __bf16* __restrict__ out, long long n)
{
  long long i = ((long long)blockIdx.x * 256 + threadIdx.x) * 8;
  if (i + 8 <= n) {
    float4 a = *(const float4*)(in + i);
    float4 b = *(const float4*)(in + i + 4);
    v8bf v;
    v[0] = (__bf16)a.x; v[1] = (__bf16)a.y; v[2] = (__bf16)a.z; v[3] = (__bf16)a.w;
    v[4] = (__bf16)b.x; v[5] = (__bf16)b.y; v[6] = (__bf16)b.z; v[7] = (__bf16)b.w;
    *(v8bf*)(out + i) = v;
  }
}

static inline int cdiv(int a, int b) { return (a + b - 1) / b; }

extern "C" void kernel_launch(void* const* d_in, const int* in_sizes, int n_in,
                              void* d_out, int out_size, void* d_ws, size_t ws_size,
                              hipStream_t stream) {
  const float* x   = (const float*)d_in[0];
  const float* adj = (const float*)d_in[1];
  const float* W1  = (const float*)d_in[2];
  const float* b1  = (const float*)d_in[3];
  const float* W2  = (const float*)d_in[4];
  const float* b2  = (const float*)d_in[5];
  const float* W3  = (const float*)d_in[6];
  const float* b3  = (const float*)d_in[7];
  const float* Wf  = (const float*)d_in[8];
  const float* bfv = (const float*)d_in[9];

  const int NN = 10000, F = 512, HD = 512, H2 = 1024, NC = 151;

  // workspace layout
  const size_t OFF_T   = 0;                       // 10000x1024 f32   40,960,000
  const size_t OFF_TT  = OFF_T  + 40960000;       // 1024x10000 bf16  20,480,000
  const size_t OFF_H   = OFF_TT + 20480000;       // 10000x1024 bf16  20,480,000
  const size_t OFF_W   = OFF_H  + 20480000;       // weights bf16      2,406,400
  const size_t OFF_XB  = OFF_W  + 2406400;        // x bf16           10,240,000
  const size_t OFF_ADJ = OFF_XB + 10240000;       // adj bf16        200,000,000
  const size_t NEED_ASYNC = OFF_ADJ + 200000000;  // ~294.6 MB

  char*   ws   = (char*)d_ws;
  float*  T    = (float*)(ws + OFF_T);
  __bf16* Tt   = (__bf16*)(ws + OFF_TT);
  __bf16* Hb   = (__bf16*)(ws + OFF_H);
  __bf16* W1t  = (__bf16*)(ws + OFF_W);
  __bf16* W2t  = W1t + 512 * 512;
  __bf16* W3t  = W2t + 512 * 512;
  __bf16* Wft  = W3t + 1024 * 512;
  __bf16* Xb   = (__bf16*)(ws + OFF_XB);
  __bf16* Adjb = (__bf16*)(ws + OFF_ADJ);

  dim3 tb(32, 8);
  dim3 gb(256);
  const bool useAsync = ws_size >= NEED_ASYNC;

  // weights -> bf16 N-major (both paths)
  transpose_f32_bf16<<<dim3(cdiv(HD, 32), cdiv(F,  32)), tb, 0, stream>>>(W1, W1t, F,  HD);
  transpose_f32_bf16<<<dim3(cdiv(HD, 32), cdiv(HD, 32)), tb, 0, stream>>>(W2, W2t, HD, HD);
  transpose_f32_bf16<<<dim3(cdiv(H2, 32), cdiv(HD, 32)), tb, 0, stream>>>(W3, W3t, HD, H2);
  transpose_f32_bf16<<<dim3(cdiv(NC, 32), cdiv(H2, 32)), tb, 0, stream>>>(Wf, Wft, H2, NC);

  if (useAsync) {
    // pre-convert to bf16 so every GEMM uses the async-DMA path
    convert_f32_bf16<<<cdiv(100000000 / 8, 256), gb, 0, stream>>>(adj, Adjb, 100000000LL);
    convert_f32_bf16<<<cdiv(5120000 / 8,   256), gb, 0, stream>>>(x,   Xb,   5120000LL);

    gemm_wmma<1,false,false,false,false><<<dim3(cdiv(HD,128), cdiv(NN,128)), gb, 0, stream>>>(
        Xb, W1t, nullptr, T, NN, HD, F, F, F);
    transpose_f32_bf16<<<dim3(cdiv(HD, 32), cdiv(NN, 32)), tb, 0, stream>>>(T, Tt, NN, HD);
    gemm_wmma<1,true,true,true,true><<<dim3(cdiv(HD,128), cdiv(NN,128)), gb, 0, stream>>>(
        Adjb, Tt, b1, Hb, NN, HD, NN, NN, NN);

    gemm_wmma<1,false,false,false,false><<<dim3(cdiv(HD,128), cdiv(NN,128)), gb, 0, stream>>>(
        Hb, W2t, nullptr, T, NN, HD, HD, HD, HD);
    transpose_f32_bf16<<<dim3(cdiv(HD, 32), cdiv(NN, 32)), tb, 0, stream>>>(T, Tt, NN, HD);
    gemm_wmma<1,true,true,true,true><<<dim3(cdiv(HD,128), cdiv(NN,128)), gb, 0, stream>>>(
        Adjb, Tt, b2, Hb, NN, HD, NN, NN, NN);

    gemm_wmma<1,false,false,false,false><<<dim3(cdiv(H2,128), cdiv(NN,128)), gb, 0, stream>>>(
        Hb, W3t, nullptr, T, NN, H2, HD, HD, HD);
    transpose_f32_bf16<<<dim3(cdiv(H2, 32), cdiv(NN, 32)), tb, 0, stream>>>(T, Tt, NN, H2);
    gemm_wmma<1,true,true,true,true><<<dim3(cdiv(H2,128), cdiv(NN,128)), gb, 0, stream>>>(
        Adjb, Tt, b3, Hb, NN, H2, NN, NN, NN);

    gemm_wmma<1,false,true,false,false><<<dim3(cdiv(NC,128), cdiv(NN,128)), gb, 0, stream>>>(
        Hb, Wft, bfv, d_out, NN, NC, H2, H2, H2);
  } else {
    // fallback: adj/x consumed as f32 with in-kernel conversion
    gemm_wmma<0,false,false,false,false><<<dim3(cdiv(HD,128), cdiv(NN,128)), gb, 0, stream>>>(
        x, W1t, nullptr, T, NN, HD, F, F, F);
    transpose_f32_bf16<<<dim3(cdiv(HD, 32), cdiv(NN, 32)), tb, 0, stream>>>(T, Tt, NN, HD);
    gemm_wmma<0,true,true,true,false><<<dim3(cdiv(HD,128), cdiv(NN,128)), gb, 0, stream>>>(
        adj, Tt, b1, Hb, NN, HD, NN, NN, NN);

    gemm_wmma<1,false,false,false,false><<<dim3(cdiv(HD,128), cdiv(NN,128)), gb, 0, stream>>>(
        Hb, W2t, nullptr, T, NN, HD, HD, HD, HD);
    transpose_f32_bf16<<<dim3(cdiv(HD, 32), cdiv(NN, 32)), tb, 0, stream>>>(T, Tt, NN, HD);
    gemm_wmma<0,true,true,true,false><<<dim3(cdiv(HD,128), cdiv(NN,128)), gb, 0, stream>>>(
        adj, Tt, b2, Hb, NN, HD, NN, NN, NN);

    gemm_wmma<1,false,false,false,false><<<dim3(cdiv(H2,128), cdiv(NN,128)), gb, 0, stream>>>(
        Hb, W3t, nullptr, T, NN, H2, HD, HD, HD);
    transpose_f32_bf16<<<dim3(cdiv(H2, 32), cdiv(NN, 32)), tb, 0, stream>>>(T, Tt, NN, H2);
    gemm_wmma<0,true,true,true,false><<<dim3(cdiv(H2,128), cdiv(NN,128)), gb, 0, stream>>>(
        adj, Tt, b3, Hb, NN, H2, NN, NN, NN);

    gemm_wmma<1,false,true,false,false><<<dim3(cdiv(NC,128), cdiv(NN,128)), gb, 0, stream>>>(
        Hb, Wft, bfv, d_out, NN, NC, H2, H2, H2);
  }
}
